// GCN_40037685134216
// MI455X (gfx1250) — compile-verified, matrix operand order checked by reading, not run on hardware
//
#include <hip/hip_runtime.h>
#include <hip/hip_bf16.h>

#define N_NODES  100000
#define N_EDGES  1600000
#define D        128
#define N_LAYERS 4
#define N_GRAPHS 512

typedef __attribute__((ext_vector_type(16))) _Float16 v16h;
typedef __attribute__((ext_vector_type(8)))  _Float16 v8h;
typedef __attribute__((ext_vector_type(8)))  float    v8f;

// ---------------------------------------------------------------------------
// small elementwise helpers
// ---------------------------------------------------------------------------
__global__ void k_fill(float* __restrict__ p, float v, int n) {
    int i = blockIdx.x * blockDim.x + threadIdx.x;
    if (i < n) p[i] = v;
}

__global__ void k_deg_scatter(const long long* __restrict__ ecol,
                              float* __restrict__ deg, int ne) {
    int e = blockIdx.x * blockDim.x + threadIdx.x;
    if (e < ne) atomicAdd(&deg[(int)ecol[e]], 1.0f);
}

__global__ void k_dinv(float* __restrict__ d, int n) {
    int i = blockIdx.x * blockDim.x + threadIdx.x;
    if (i < n) { float v = d[i]; d[i] = (v > 0.0f) ? rsqrtf(v) : 0.0f; }
}

__global__ void k_norm(const long long* __restrict__ erow,
                       const long long* __restrict__ ecol,
                       const float* __restrict__ dinv,
                       float* __restrict__ norm, int ne) {
    int e = blockIdx.x * blockDim.x + threadIdx.x;
    if (e < ne) norm[e] = dinv[(int)erow[e]] * dinv[(int)ecol[e]];
}

// ---------------------------------------------------------------------------
// Fused GEMM + self-loop epilogue:
//   xw[r,:]   = (relu_in ? relu(Hin[r,:]) : Hin[r,:]) @ W
//   outA[r,:] = bias + dinv[r]^2 * xw[r,:]        (scatter accumulator init)
// Row-local => safe with Hin == outA (rows staged to LDS before stores).
// 64 rows/block, 4 waves; wave: 1 Mtile x 8 Ntiles, K = 4 x 32 (f16, f32 acc).
// B fragments batched per N-tile so ds_load_b128s overlap the WMMA chain.
// C staged through LDS for coalesced float4 global stores.
// ---------------------------------------------------------------------------
#define GROWS   64
#define GPAD    8                      // +16B row pad (f16) -> rotates banks
#define APITCH  (D + GPAD)             // 136 halves
#define CPITCH  (D + 4)                // 132 floats: half-waves on disjoint banks
#define SA_BYTES  (GROWS * APITCH * 2)         // 17408
#define SW_BYTES  (D * APITCH * 2)             // 34816
#define SC_BYTES  (GROWS * CPITCH * 4)         // 33792
#define SMEM_BYTES (SA_BYTES + SW_BYTES)       // 52224 (>= SC_BYTES, aliased)

__global__ __launch_bounds__(128)
void k_gemm_wmma(const float* Hin, const float* __restrict__ W,
                 float* __restrict__ Xw, const float* __restrict__ dinv,
                 const float* __restrict__ bias, float* OutA,
                 int relu_in, int nrows) {
    __shared__ __align__(16) unsigned char smem[SMEM_BYTES];
    _Float16* sA = (_Float16*)smem;                  // [GROWS][APITCH]
    _Float16* sW = (_Float16*)(smem + SA_BYTES);     // [D][APITCH], sW[n][k]=W[k][n]
    float*    sC = (float*)smem;                     // [GROWS][CPITCH] (aliased)

    const int tid  = threadIdx.x;
    const int row0 = blockIdx.x * GROWS;

    // stage W transposed (f32 -> f16)
    for (int i = tid; i < D * D; i += 128) {
        int k = i >> 7, n = i & (D - 1);
        sW[n * APITCH + k] = (_Float16)W[i];
    }
    // stage h tile (fused relu of previous layer, zero-pad OOB rows)
    for (int i = tid; i < GROWS * D; i += 128) {
        int r = i >> 7, c = i & (D - 1);
        int gr = row0 + r;
        float v = 0.0f;
        if (gr < nrows) {
            v = Hin[(size_t)gr * D + c];
            if (relu_in) v = fmaxf(v, 0.0f);
        }
        sA[r * APITCH + c] = (_Float16)v;
    }
    __syncthreads();

    const int wave = tid >> 5;
    const int lane = tid & 31;
    const int hsel = lane >> 4;       // 0: K pattern base, 1: +8
    const int l16  = lane & 15;
    const int m0   = wave * 16;

    // per-lane A fragments for the 4 K-steps (ISA 7.12.2 16-bit A layout)
    v16h afrag[4];
#pragma unroll
    for (int ks = 0; ks < 4; ++ks) {
        int kb = ks * 32 + hsel * 8;
        const _Float16* ap = sA + (m0 + l16) * APITCH;
        v8h lo = *(const v8h*)(ap + kb);
        v8h hi = *(const v8h*)(ap + kb + 16);
        v16h a;
#pragma unroll
        for (int j = 0; j < 8; ++j) { a[j] = lo[j]; a[j + 8] = hi[j]; }
        afrag[ks] = a;
    }

    // 8 N-tiles, accumulate in registers; batch all 4 B frags per tile so the
    // 8 ds_load_b128s are in flight together and dscnt waits stagger.
    v8f acc[8];
#pragma unroll
    for (int nt = 0; nt < 8; ++nt) {
        int n0 = nt * 16;
        const _Float16* bp = sW + (n0 + l16) * APITCH;
        v16h bfrag[4];
#pragma unroll
        for (int ks = 0; ks < 4; ++ks) {
            int kb = ks * 32 + hsel * 8;
            v8h lo = *(const v8h*)(bp + kb);
            v8h hi = *(const v8h*)(bp + kb + 16);
            v16h b;
#pragma unroll
            for (int j = 0; j < 8; ++j) { b[j] = lo[j]; b[j + 8] = hi[j]; }
            bfrag[ks] = b;
        }
        v8f c = {};
#pragma unroll
        for (int ks = 0; ks < 4; ++ks) {
            c = __builtin_amdgcn_wmma_f32_16x16x32_f16(
                false, afrag[ks], false, bfrag[ks], (short)0, c, false, false);
        }
        acc[nt] = c;
    }
    __syncthreads();   // sA/sW dead; safe to alias as sC

    // dump C tiles to LDS (C/D layout: vgpr v -> row m0 + hsel*8 + v, col n0+l16)
#pragma unroll
    for (int nt = 0; nt < 8; ++nt) {
#pragma unroll
        for (int v = 0; v < 8; ++v) {
            sC[(m0 + hsel * 8 + v) * CPITCH + nt * 16 + l16] = acc[nt][v];
        }
    }
    __syncthreads();

    // coalesced float4 stores: xw and fused self-loop/bias accumulator init
    for (int i = tid; i < GROWS * (D / 4); i += 128) {
        int r  = i >> 5;          // D/4 == 32 chunks per row
        int c4 = i & 31;
        int gr = row0 + r;
        if (gr >= nrows) continue;
        float4 v = ((const float4*)(sC + r * CPITCH))[c4];
        ((float4*)(Xw + (size_t)gr * D))[c4] = v;
        float s = dinv[gr]; s *= s;
        float4 b = ((const float4*)bias)[c4];
        float4 o = { b.x + s * v.x, b.y + s * v.y, b.z + s * v.z, b.w + s * v.w };
        ((float4*)(OutA + (size_t)gr * D))[c4] = o;
    }
}

// one wave per edge: out[col,:] += norm[e] * xw[row,:]
__global__ __launch_bounds__(256)
void k_edge_scatter(const float* __restrict__ xw, const long long* __restrict__ erow,
                    const long long* __restrict__ ecol, const float* __restrict__ norm,
                    float* __restrict__ out, int ne) {
    int g = blockIdx.x * blockDim.x + threadIdx.x;
    int e = g >> 5, lane = g & 31;
    if (e >= ne) return;
    int r = (int)erow[e], c = (int)ecol[e];
    float nv = norm[e];
    float4 v = ((const float4*)(xw + (size_t)r * D))[lane];
    float* dst = out + (size_t)c * D + lane * 4;
    atomicAdd(dst + 0, v.x * nv);
    atomicAdd(dst + 1, v.y * nv);
    atomicAdd(dst + 2, v.z * nv);
    atomicAdd(dst + 3, v.w * nv);
}

// one wave per node: pooled[batch[n],:] += h[n,:]; counts[batch[n]] += 1
__global__ __launch_bounds__(256)
void k_pool_sum(const float* __restrict__ h, const long long* __restrict__ batch,
                float* __restrict__ pooled, float* __restrict__ counts, int nn) {
    int g = blockIdx.x * blockDim.x + threadIdx.x;
    int n = g >> 5, lane = g & 31;
    if (n >= nn) return;
    int gr = (int)batch[n];
    float4 v = ((const float4*)(h + (size_t)n * D))[lane];
    float* dst = pooled + (size_t)gr * D + lane * 4;
    atomicAdd(dst + 0, v.x);
    atomicAdd(dst + 1, v.y);
    atomicAdd(dst + 2, v.z);
    atomicAdd(dst + 3, v.w);
    if (lane == 0) atomicAdd(&counts[gr], 1.0f);
}

// one block (128 thr) per graph: mean, 128->100 relu, 100->4
__global__ __launch_bounds__(128)
void k_mlp(const float* __restrict__ pooled, const float* __restrict__ counts,
           const float* __restrict__ w1, const float* __restrict__ b1,
           const float* __restrict__ w2, const float* __restrict__ b2,
           float* __restrict__ pred) {
    __shared__ float sp[D];
    __shared__ float sh[100];
    int g = blockIdx.x, t = threadIdx.x;
    float cnt = fmaxf(counts[g], 1.0f);
    sp[t] = pooled[(size_t)g * D + t] / cnt;
    __syncthreads();
    if (t < 100) {
        float acc = b1[t];
        for (int d = 0; d < D; ++d) acc += sp[d] * w1[d * 100 + t];
        sh[t] = fmaxf(acc, 0.0f);
    }
    __syncthreads();
    if (t < 4) {
        float acc = b2[t];
        for (int j = 0; j < 100; ++j) acc += sh[j] * w2[j * 4 + t];
        pred[g * 4 + t] = acc;
    }
}

// ---------------------------------------------------------------------------
extern "C" void kernel_launch(void* const* d_in, const int* in_sizes, int n_in,
                              void* d_out, int out_size, void* d_ws, size_t ws_size,
                              hipStream_t stream) {
    const float*     x   = (const float*)d_in[0];
    const float*     Ws  = (const float*)d_in[1];
    const float*     bs  = (const float*)d_in[2];
    const float*     w1  = (const float*)d_in[3];
    const float*     b1  = (const float*)d_in[4];
    const float*     w2  = (const float*)d_in[5];
    const float*     b2  = (const float*)d_in[6];
    const long long* ei  = (const long long*)d_in[7];
    const long long* bat = (const long long*)d_in[8];
    const long long* erow = ei;             // sources
    const long long* ecol = ei + N_EDGES;   // targets
    float* pred = (float*)d_out;

    // workspace carve-up (256B aligned)
    char* wp = (char*)d_ws;
    size_t off = 0;
    auto take = [&](size_t bytes) {
        void* p = wp + off;
        off += (bytes + 255) & ~(size_t)255;
        return p;
    };
    float* bufA   = (float*)take((size_t)N_NODES * D * 4);   // h / accumulator
    float* bufB   = (float*)take((size_t)N_NODES * D * 4);   // xw
    float* dinv   = (float*)take((size_t)N_NODES * 4);
    float* norm   = (float*)take((size_t)N_EDGES * 4);
    float* pooled = (float*)take((size_t)N_GRAPHS * D * 4);
    float* counts = (float*)take((size_t)N_GRAPHS * 4);
    (void)ws_size; (void)in_sizes; (void)n_in; (void)out_size;

    // 1) degrees (self-loop => init 1), dinv, per-edge norm
    k_fill<<<(N_NODES + 255) / 256, 256, 0, stream>>>(dinv, 1.0f, N_NODES);
    k_deg_scatter<<<(N_EDGES + 255) / 256, 256, 0, stream>>>(ecol, dinv, N_EDGES);
    k_dinv<<<(N_NODES + 255) / 256, 256, 0, stream>>>(dinv, N_NODES);
    k_norm<<<(N_EDGES + 255) / 256, 256, 0, stream>>>(erow, ecol, dinv, norm, N_EDGES);

    // 2) GCN layers: fused (GEMM + bias + self-loop init), then edge scatter
    const int gemm_blocks  = (N_NODES + GROWS - 1) / GROWS;
    const int node_wave_bl = (N_NODES * 32 + 255) / 256;
    const int edge_wave_bl = (N_EDGES * 32 + 255) / 256;
    const float* hin = x;
    for (int l = 0; l < N_LAYERS; ++l) {
        k_gemm_wmma<<<gemm_blocks, 128, 0, stream>>>(
            hin, Ws + (size_t)l * D * D, bufB, dinv, bs + (size_t)l * D, bufA,
            (l > 0) ? 1 : 0, N_NODES);
        k_edge_scatter<<<edge_wave_bl, 256, 0, stream>>>(
            bufB, erow, ecol, norm, bufA, N_EDGES);
        hin = bufA;
    }

    // 3) mean pool + MLP head
    k_fill<<<(N_GRAPHS * D + 255) / 256, 256, 0, stream>>>(pooled, 0.0f, N_GRAPHS * D);
    k_fill<<<(N_GRAPHS + 255) / 256, 256, 0, stream>>>(counts, 0.0f, N_GRAPHS);
    k_pool_sum<<<node_wave_bl, 256, 0, stream>>>(bufA, bat, pooled, counts, N_NODES);
    k_mlp<<<N_GRAPHS, 128, 0, stream>>>(pooled, counts, w1, b1, w2, b2, pred);
}
